// MultiheadAttention_66151086293471
// MI455X (gfx1250) — compile-verified, matrix operand order checked by reading, not run on hardware
//
#include <hip/hip_runtime.h>
#include <hip/hip_bf16.h>

// MI455X (gfx1250) fused multi-head attention, bf16 WMMA path.
// fp32 inputs; projections write bf16 copies laid out for direct WMMA
// fragment loads (row-major for Q/K, transposed for V) so the attention
// inner loop is pure v8bf loads + v_wmma_f32_16x16x32_bf16.

#define D_MODEL 1024
#define NHEADS  16
#define DKH     64
#define SEQ     2048
#define BATCH   2

typedef __bf16 bf16_t;
typedef __attribute__((ext_vector_type(16))) __bf16 v16bf;
typedef __attribute__((ext_vector_type(8)))  __bf16 v8bf;
typedef __attribute__((ext_vector_type(8)))  float  v8f;
typedef __attribute__((ext_vector_type(4)))  float  f32x4;

#define WMMA_BF16(A, B, C) \
  __builtin_amdgcn_wmma_f32_16x16x32_bf16(false, (A), false, (B), (short)0, (C), false, false)

// v16bf fragment from two runs of 8 consecutive fp32 (with cvt).
// ISA 16-bit A/B layout: lane<16 holds K {0..7,16..23}, lane>=16 {8..15,24..31}.
__device__ __forceinline__ v16bf frag_f32pair(const float* p0, const float* p1) {
  f32x4 a = *(const f32x4*)p0;
  f32x4 b = *(const f32x4*)(p0 + 4);
  f32x4 c = *(const f32x4*)p1;
  f32x4 d = *(const f32x4*)(p1 + 4);
  v16bf r;
  r[0]=(bf16_t)a[0];  r[1]=(bf16_t)a[1];  r[2]=(bf16_t)a[2];  r[3]=(bf16_t)a[3];
  r[4]=(bf16_t)b[0];  r[5]=(bf16_t)b[1];  r[6]=(bf16_t)b[2];  r[7]=(bf16_t)b[3];
  r[8]=(bf16_t)c[0];  r[9]=(bf16_t)c[1];  r[10]=(bf16_t)c[2]; r[11]=(bf16_t)c[3];
  r[12]=(bf16_t)d[0]; r[13]=(bf16_t)d[1]; r[14]=(bf16_t)d[2]; r[15]=(bf16_t)d[3];
  return r;
}

// v16bf fragment from two 16-byte bf16 runs (no conversion).
__device__ __forceinline__ v16bf frag_bf(const bf16_t* p0, const bf16_t* p1) {
  v8bf lo = *(const v8bf*)p0;
  v8bf hi = *(const v8bf*)p1;
  v16bf r;
#pragma unroll
  for (int e = 0; e < 8; ++e) { r[e] = lo[e]; r[8 + e] = hi[e]; }
  return r;
}

// C = (A[M,K] @ W[N,K]^T + bias[N]) * scale.
// One wave computes a 16x64 strip (4 N-tiles share one A fragment).
// Optional outputs: Cf (fp32 row-major), Cbf (bf16 row-major),
// CbfT (bf16 transposed [N][M] — contiguous 8-elem packed store per lane).
__global__ __launch_bounds__(128) void gemm_wtb(
    const float* __restrict__ A, const float* __restrict__ W,
    const float* __restrict__ bias,
    float* __restrict__ Cf, bf16_t* __restrict__ Cbf, bf16_t* __restrict__ CbfT,
    int M, int N, int K, float scale) {
  const int wib    = threadIdx.x >> 5;
  const int wave   = blockIdx.x * 4 + wib;
  const int nstrip = N >> 6;
  const int tm     = wave / nstrip;
  const int col0   = (wave % nstrip) * 64;
  if (tm * 16 >= M) return;
  const int lane = threadIdx.x & 31;
  const int grp  = lane >> 4;   // K-half selector
  const int idx  = lane & 15;   // row (A) / column (B,C)

  const float* arow = A + (size_t)(tm * 16 + idx) * K;
  const float* wrow0 = W + (size_t)(col0 + 0  + idx) * K;
  const float* wrow1 = W + (size_t)(col0 + 16 + idx) * K;
  const float* wrow2 = W + (size_t)(col0 + 32 + idx) * K;
  const float* wrow3 = W + (size_t)(col0 + 48 + idx) * K;

  v8f acc0 = {}, acc1 = {}, acc2 = {}, acc3 = {};
  for (int k0 = 0; k0 < K; k0 += 32) {
    v16bf af = frag_f32pair(arow + k0 + grp * 8, arow + k0 + 16 + grp * 8);
    acc0 = WMMA_BF16(af, frag_f32pair(wrow0 + k0 + grp * 8, wrow0 + k0 + 16 + grp * 8), acc0);
    acc1 = WMMA_BF16(af, frag_f32pair(wrow1 + k0 + grp * 8, wrow1 + k0 + 16 + grp * 8), acc1);
    acc2 = WMMA_BF16(af, frag_f32pair(wrow2 + k0 + grp * 8, wrow2 + k0 + 16 + grp * 8), acc2);
    acc3 = WMMA_BF16(af, frag_f32pair(wrow3 + k0 + grp * 8, wrow3 + k0 + 16 + grp * 8), acc3);
  }

  v8f* accs[4] = {&acc0, &acc1, &acc2, &acc3};
#pragma unroll
  for (int t = 0; t < 4; ++t) {
    const int  col = col0 + t * 16 + idx;
    const float bn = bias[col];
    float vals[8];
#pragma unroll
    for (int r = 0; r < 8; ++r) vals[r] = ((*accs[t])[r] + bn) * scale;
    if (Cf) {
#pragma unroll
      for (int r = 0; r < 8; ++r)
        Cf[(size_t)(tm * 16 + r + grp * 8) * N + col] = vals[r];
    }
    if (Cbf) {
#pragma unroll
      for (int r = 0; r < 8; ++r)
        Cbf[(size_t)(tm * 16 + r + grp * 8) * N + col] = (bf16_t)vals[r];
    }
    if (CbfT) {  // transposed: rows r are contiguous M -> one 16B packed store
      v8bf pk;
#pragma unroll
      for (int r = 0; r < 8; ++r) pk[r] = (bf16_t)vals[r];
      *(v8bf*)(CbfT + (size_t)col * M + tm * 16 + grp * 8) = pk;
    }
  }
}

// Flash-attention: one wave per (b, h, 16-row q tile).
// qbf: scaled Q projection, bf16 row-major [B*L][D].
// kbf: K projection, bf16 row-major [B*L][D].
// vpT: V projection, bf16 transposed [D][B*L].
__global__ __launch_bounds__(128) void attn_fused(
    const bf16_t* __restrict__ qbf, const bf16_t* __restrict__ kbf,
    const bf16_t* __restrict__ vpT, const float* __restrict__ q_mask,
    const float* __restrict__ k_mask, const float* __restrict__ pos,
    float* __restrict__ out) {
  __shared__ bf16_t plds_all[4][16 * 32];  // per-wave P staging (C->A relayout)
  const int wib  = threadIdx.x >> 5;
  bf16_t* plds   = &plds_all[wib][0];
  const int wave = blockIdx.x * 4 + wib;
  const int qt   = wave & 127;
  const int h    = (wave >> 7) & 15;
  const int b    = wave >> 11;
  const int lane = threadIdx.x & 31;
  const int grp  = lane >> 4;
  const int idx  = lane & 15;
  const int q0   = qt * 16;
  const int M    = BATCH * SEQ;

  // Q fragments (already scaled by dkh^-0.5 in projection)
  const bf16_t* qb = qbf + ((size_t)(b * SEQ) + q0 + idx) * D_MODEL + h * DKH;
  v16bf qa0 = frag_bf(qb + grp * 8,      qb + 16 + grp * 8);
  v16bf qa1 = frag_bf(qb + 32 + grp * 8, qb + 48 + grp * 8);

  float mrow[8], lrow[8], qmrow[8];
  v8f o0 = {}, o1 = {}, o2 = {}, o3 = {};
#pragma unroll
  for (int r = 0; r < 8; ++r) {
    mrow[r] = -1e30f;
    lrow[r] = 0.f;
    qmrow[r] = q_mask[b * SEQ + q0 + r + grp * 8];
  }

  for (int j0 = 0; j0 < SEQ; j0 += 32) {
    // ---- S = Q*K^T for keys j0..j0+31 ----
    const bf16_t* k0b = kbf + ((size_t)(b * SEQ) + j0 + idx) * D_MODEL + h * DKH;
    const bf16_t* k1b = k0b + 16 * D_MODEL;
    v8f s0 = {}, s1 = {};
    s0 = WMMA_BF16(qa0, frag_bf(k0b + grp * 8,      k0b + 16 + grp * 8), s0);
    s0 = WMMA_BF16(qa1, frag_bf(k0b + 32 + grp * 8, k0b + 48 + grp * 8), s0);
    s1 = WMMA_BF16(qa0, frag_bf(k1b + grp * 8,      k1b + 16 + grp * 8), s1);
    s1 = WMMA_BF16(qa1, frag_bf(k1b + 32 + grp * 8, k1b + 48 + grp * 8), s1);

    if (j0 + 32 < SEQ)  // gfx1250 global_prefetch_b8 on next K tile
      __builtin_prefetch(kbf + ((size_t)(b * SEQ) + j0 + 32 + idx) * D_MODEL + h * DKH, 0, 1);

    const float km0 = k_mask[b * SEQ + j0 + idx];
    const float km1 = k_mask[b * SEQ + j0 + 16 + idx];

    // ---- add pos_logits + additive mask; row max across 32 logits ----
    float rmax[8];
#pragma unroll
    for (int r = 0; r < 8; ++r) {
      const int qrow = q0 + r + grp * 8;
      const float* pr = pos + ((size_t)(b * SEQ) + qrow) * SEQ + j0;
      float a0 = (qmrow[r] * km0 == 0.f) ? -1e30f : 0.f;
      float a1 = (qmrow[r] * km1 == 0.f) ? -1e30f : 0.f;
      s0[r] += pr[idx] + a0;
      s1[r] += pr[16 + idx] + a1;
      float v = fmaxf(s0[r], s1[r]);
      v = fmaxf(v, __shfl_xor(v, 1, 32));   // rows live within 16-lane halves
      v = fmaxf(v, __shfl_xor(v, 2, 32));
      v = fmaxf(v, __shfl_xor(v, 4, 32));
      v = fmaxf(v, __shfl_xor(v, 8, 32));
      rmax[r] = v;
    }

    // ---- online softmax update; stage P (bf16) to LDS ----
#pragma unroll
    for (int r = 0; r < 8; ++r) {
      float mn   = fmaxf(mrow[r], rmax[r]);
      float corr = __expf(mrow[r] - mn);
      mrow[r] = mn;
      float p0 = __expf(s0[r] - mn);
      float p1 = __expf(s1[r] - mn);
      int m = r + grp * 8;
      plds[m * 32 + idx]      = (bf16_t)p0;
      plds[m * 32 + 16 + idx] = (bf16_t)p1;
      float rs = p0 + p1;
      rs += __shfl_xor(rs, 1, 32);
      rs += __shfl_xor(rs, 2, 32);
      rs += __shfl_xor(rs, 4, 32);
      rs += __shfl_xor(rs, 8, 32);
      lrow[r] = lrow[r] * corr + rs;
      o0[r] *= corr; o1[r] *= corr; o2[r] *= corr; o3[r] *= corr;
    }

    // ---- P as A-fragment (16x32) from wave-private LDS ----
    v16bf pf = frag_bf(plds + idx * 32 + grp * 8, plds + idx * 32 + 16 + grp * 8);

    // ---- O += P * V: vpT gives contiguous K-runs per column ----
    const bf16_t* vb = vpT + (size_t)(h * DKH + idx) * M + b * SEQ + j0;
    const size_t vcs = (size_t)16 * M;  // column stride between tiles
    o0 = WMMA_BF16(pf, frag_bf(vb + 0 * vcs + grp * 8, vb + 0 * vcs + 16 + grp * 8), o0);
    o1 = WMMA_BF16(pf, frag_bf(vb + 1 * vcs + grp * 8, vb + 1 * vcs + 16 + grp * 8), o1);
    o2 = WMMA_BF16(pf, frag_bf(vb + 2 * vcs + grp * 8, vb + 2 * vcs + 16 + grp * 8), o2);
    o3 = WMMA_BF16(pf, frag_bf(vb + 3 * vcs + grp * 8, vb + 3 * vcs + 16 + grp * 8), o3);
  }

  // ---- normalize, write combined-head layout [B, L, Nh*dvh] ----
#pragma unroll
  for (int r = 0; r < 8; ++r) {
    float inv = lrow[r] > 0.f ? 1.f / lrow[r] : 0.f;
    const int qrow = q0 + r + grp * 8;
    float* orow = out + ((size_t)(b * SEQ) + qrow) * D_MODEL + h * DKH;
    orow[idx]      = o0[r] * inv;
    orow[16 + idx] = o1[r] * inv;
    orow[32 + idx] = o2[r] * inv;
    orow[48 + idx] = o3[r] * inv;
  }
}

extern "C" void kernel_launch(void* const* d_in, const int* in_sizes, int n_in,
                              void* d_out, int out_size, void* d_ws, size_t ws_size,
                              hipStream_t stream) {
  const float* q   = (const float*)d_in[0];
  const float* k   = (const float*)d_in[1];
  const float* qm  = (const float*)d_in[2];
  const float* km  = (const float*)d_in[3];
  const float* pos = (const float*)d_in[4];
  const float* Wq  = (const float*)d_in[5];
  const float* bq  = (const float*)d_in[6];
  const float* Wk  = (const float*)d_in[7];
  const float* bk  = (const float*)d_in[8];
  const float* Wv  = (const float*)d_in[9];
  const float* bv  = (const float*)d_in[10];
  const float* Wo  = (const float*)d_in[11];
  const float* bo  = (const float*)d_in[12];
  float* out = (float*)d_out;

  const int M = BATCH * SEQ;                 // 4096
  const size_t plane = (size_t)M * D_MODEL;  // 4M elements

  char* w = (char*)d_ws;
  bf16_t* qbf = (bf16_t*)w; w += plane * sizeof(bf16_t);  // 8 MB  scaled Q, bf16 row-major
  float*  kpf = (float*)w;  w += plane * sizeof(float);   // 16 MB K proj fp32 (input to V proj)
  bf16_t* kbf = (bf16_t*)w; w += plane * sizeof(bf16_t);  // 8 MB  K proj bf16 row-major
  bf16_t* vpT = (bf16_t*)w; w += plane * sizeof(bf16_t);  // 8 MB  V proj bf16 transposed [D][M]
  float*  aw  = (float*)w;                                // 16 MB attention out (pre-Wo)

  dim3 blk(128);
  const int gemm_blocks = (M / 16) * (D_MODEL / 64) / 4;     // 1024 (16x64 strips)
  const int attn_blocks = (BATCH * NHEADS * (SEQ / 16)) / 4; // 1024

  hipLaunchKernelGGL(gemm_wtb, dim3(gemm_blocks), blk, 0, stream,
                     q, Wq, bq, (float*)nullptr, qbf, (bf16_t*)nullptr,
                     M, D_MODEL, D_MODEL, 0.125f);           // dkh^-0.5
  hipLaunchKernelGGL(gemm_wtb, dim3(gemm_blocks), blk, 0, stream,
                     k, Wk, bk, kpf, kbf, (bf16_t*)nullptr,
                     M, D_MODEL, D_MODEL, 1.0f);
  hipLaunchKernelGGL(gemm_wtb, dim3(gemm_blocks), blk, 0, stream,
                     kpf, Wv, bv, (float*)nullptr, (bf16_t*)nullptr, vpT,
                     M, D_MODEL, D_MODEL, 1.0f);
  hipLaunchKernelGGL(attn_fused, dim3(attn_blocks), blk, 0, stream,
                     qbf, kbf, vpT, qm, km, pos, aw);
  hipLaunchKernelGGL(gemm_wtb, dim3(gemm_blocks), blk, 0, stream,
                     aw, Wo, bo, out, (bf16_t*)nullptr, (bf16_t*)nullptr,
                     M, D_MODEL, D_MODEL, 1.0f);
}
